// BoxTransformer_420906795019
// MI455X (gfx1250) — matrix operand order repaired and checked by reading, not compiled, not run on hardware
//
#include <hip/hip_runtime.h>
#include <hip/hip_bf16.h>
#include <math.h>

// ---------------- model constants ----------------
constexpr int Bn  = 2;
constexpr int D   = 256;
constexpr int NH  = 8;
constexpr int DH  = 32;     // D / NH
constexpr int Lv  = 4;      // levels
constexpr int NP  = 4;      // points per level (K*K)
constexpr int AA  = NH * Lv * NP;   // 128
constexpr int FF  = 1024;
constexpr int NQ  = 300;
constexpr int NE  = 6;
constexpr int ND  = 6;
constexpr int S   = 64*64 + 32*32 + 16*16 + 8*8;   // 5440

typedef __attribute__((ext_vector_type(16))) __bf16 v16bf;
typedef __attribute__((ext_vector_type(8)))  float  v8f;

static __device__ __forceinline__ __bf16 f2bf(float x) { return (__bf16)x; }

// ======================================================================
// WMMA GEMM:  Y[M,N] = act( X[M,K] @ W[K,N] + bias[N] )
// W row stride = wld (supports column slices of a wider matrix).
// Requirements: K % 32 == 0, N % 64 == 0. M may be ragged.
// Block: 256 threads = 8 waves; block tile 128(M) x 64(N), K-step 32.
// Wave grid 4(M) x 2(N); wave tile 32x32 = 4 x v_wmma_f32_16x16x32_bf16.
// Double-buffered LDS staging (bf16): A [m][k] 128x32, B transposed [n][k]
// 64x32 so both A and B fragments load as contiguous ds_load_b128 pairs.
// ======================================================================
__global__ void __launch_bounds__(256)
k_gemm(const float* __restrict__ X, const float* __restrict__ W,
       const float* __restrict__ bias, float* __restrict__ Y,
       int M, int N, int K, int wld, int relu)
{
    __shared__ __bf16 As[2][128 * 32];   // [buf][m][k]   16 KB
    __shared__ __bf16 Bs[2][64 * 32];    // [buf][n][k]    8 KB (transposed)

    const int tid  = threadIdx.x;
    const int lane = tid & 31;
    const int wave = tid >> 5;
    const int wm   = wave & 3;        // 0..3  (M sub-tile of 32)
    const int wn   = wave >> 2;       // 0..1  (N sub-tile of 32)
    const int m0   = blockIdx.y * 128;
    const int n0   = blockIdx.x * 64;

    const int r    = lane & 15;
    const int half = lane >> 4;

    // ---- cooperative staging coordinates ----
    // A: 128 rows x 32 cols; thread: row = tid>>1, 16 contiguous cols.
    int a_row = m0 + (tid >> 1);
    if (a_row >= M) a_row = M - 1;               // clamp (masked at store)
    const int a_c0 = (tid & 1) * 16;
    // B (transposed into LDS): thread owns column n = tid&63, 8 consecutive k.
    const int b_n  = tid & 63;
    const int b_k0 = (tid >> 6) * 8;

    float areg[16];
    float breg[8];

    auto load_tiles = [&](int k0) {
        const float* asrc = X + (size_t)a_row * K + k0 + a_c0;
#pragma unroll
        for (int j = 0; j < 16; ++j) areg[j] = asrc[j];
        const float* bsrc = W + (size_t)(k0 + b_k0) * wld + n0 + b_n;
#pragma unroll
        for (int j = 0; j < 8; ++j) breg[j] = bsrc[(size_t)j * wld];
    };
    auto store_tiles = [&](int buf) {
        __bf16* adst = &As[buf][(tid >> 1) * 32 + a_c0];
#pragma unroll
        for (int j = 0; j < 16; ++j) adst[j] = f2bf(areg[j]);
        __bf16* bdst = &Bs[buf][b_n * 32 + b_k0];
#pragma unroll
        for (int j = 0; j < 8; ++j) bdst[j] = f2bf(breg[j]);
    };

    v8f acc[2][2];
#pragma unroll
    for (int i = 0; i < 2; ++i)
#pragma unroll
        for (int j = 0; j < 2; ++j)
            acc[i][j] = (v8f){0.f,0.f,0.f,0.f,0.f,0.f,0.f,0.f};

    load_tiles(0);
    store_tiles(0);
    __syncthreads();

    const int nk = K >> 5;
    for (int t = 0; t < nk; ++t) {
        const int cur = t & 1;
        const bool more = (t + 1) < nk;
        if (more) load_tiles((t + 1) * 32);      // prefetch (overlaps wmmas)

        // ---- fragments from LDS (contiguous b128 reads) ----
        v16bf a[2], b[2];
#pragma unroll
        for (int mi = 0; mi < 2; ++mi) {
            const int row = wm * 32 + mi * 16 + r;
#pragma unroll
            for (int e = 0; e < 16; ++e) {
                // A 16x32 bf16 layout: k = (e&7) + 16*(e>>3) + 8*half
                int ka = (e & 7) + 16 * (e >> 3) + 8 * half;
                a[mi][e] = As[cur][row * 32 + ka];
            }
        }
#pragma unroll
        for (int ni = 0; ni < 2; ++ni) {
            const int col = wn * 32 + ni * 16 + r;
#pragma unroll
            for (int e = 0; e < 16; ++e) {
                // B 32x16 bf16 layout: k = (e&7) + 16*(e>>3) + 8*half
                b[ni][e] = Bs[cur][col * 32 + (e & 7) + 16 * (e >> 3) + 8 * half];
            }
        }

#pragma unroll
        for (int mi = 0; mi < 2; ++mi)
#pragma unroll
            for (int ni = 0; ni < 2; ++ni)
                acc[mi][ni] = __builtin_amdgcn_wmma_f32_16x16x32_bf16(
                    false, a[mi], false, b[ni], (short)0, acc[mi][ni], false, false);

        if (more) {
            store_tiles(cur ^ 1);                // fill the other buffer
            __syncthreads();
        }
    }

    // ---- epilogue ----
#pragma unroll
    for (int mi = 0; mi < 2; ++mi) {
#pragma unroll
        for (int ni = 0; ni < 2; ++ni) {
            const int n  = n0 + wn * 32 + ni * 16 + r;
            const float bn = bias[n];
#pragma unroll
            for (int i = 0; i < 8; ++i) {
                int m = m0 + wm * 32 + mi * 16 + i + 8 * half;
                if (m < M) {
                    float v = acc[mi][ni][i] + bn;
                    if (relu) v = fmaxf(v, 0.f);
                    Y[(size_t)m * N + n] = v;
                }
            }
        }
    }
}

// ---------------- elementwise add ----------------
__global__ void k_add(const float* __restrict__ a, const float* __restrict__ b,
                      float* __restrict__ y, int n)
{
    int t = blockIdx.x * blockDim.x + threadIdx.x;
    if (t < n) y[t] = a[t] + b[t];
}

// ---------------- residual + layernorm over D=256 ----------------
__global__ void k_add_ln(const float* __restrict__ res, const float* __restrict__ y,
                         const float* __restrict__ g, const float* __restrict__ bta,
                         float* __restrict__ out)
{
    __shared__ float red[256];
    const size_t row = blockIdx.x;
    const int t = threadIdx.x;
    float v = y[row * 256 + t];
    if (res) v += res[row * 256 + t];
    red[t] = v; __syncthreads();
    for (int s = 128; s > 0; s >>= 1) { if (t < s) red[t] += red[t + s]; __syncthreads(); }
    float mean = red[0] / 256.f; __syncthreads();
    float d = v - mean;
    red[t] = d * d; __syncthreads();
    for (int s = 128; s > 0; s >>= 1) { if (t < s) red[t] += red[t + s]; __syncthreads(); }
    float var = red[0] / 256.f;
    out[row * 256 + t] = d * rsqrtf(var + 1e-5f) * g[t] + bta[t];
}

// ---------------- reference windows (shared across batch) ----------------
__global__ void k_refwin(float* __restrict__ refw)
{
    int s = blockIdx.x * blockDim.x + threadIdx.x;
    if (s >= S) return;
    int Hl, Wl, loc;
    if      (s < 4096) { Hl = Wl = 64; loc = s; }
    else if (s < 5120) { Hl = Wl = 32; loc = s - 4096; }
    else if (s < 5376) { Hl = Wl = 16; loc = s - 5120; }
    else               { Hl = Wl = 8;  loc = s - 5376; }
    int i = loc / Wl, j = loc % Wl;
    refw[s*4+0] = (j + 0.5f) / (Wl + 1e-6f);
    refw[s*4+1] = (i + 0.5f) / (Hl + 1e-6f);
    refw[s*4+2] = 4.0f / Wl;
    refw[s*4+3] = 4.0f / Hl;
}

// ---------------- softmax over groups of 16 (per head) ----------------
__global__ void k_softmax16(float* __restrict__ att, int groups)
{
    int t = blockIdx.x * blockDim.x + threadIdx.x;
    if (t >= groups) return;
    float* p = att + (size_t)t * 16;
    float mx = p[0];
#pragma unroll
    for (int i = 1; i < 16; ++i) mx = fmaxf(mx, p[i]);
    float e[16], s = 0.f;
#pragma unroll
    for (int i = 0; i < 16; ++i) { e[i] = expf(p[i] - mx); s += e[i]; }
    float inv = 1.f / s;
#pragma unroll
    for (int i = 0; i < 16; ++i) p[i] = e[i] * inv;
}

// ---------------- box attention sampling ----------------
// V:[Bn,S,NH,DH]  att:[Bn,Nq,NH,16]  refw:[(1|Bn),Nq,4]  out:[Bn,Nq,NH,DH]
__global__ void k_boxsample(const float* __restrict__ V, const float* __restrict__ att,
                            const float* __restrict__ refw, int refw_bstride,
                            float* __restrict__ out, int Nq)
{
    int tid = blockIdx.x * blockDim.x + threadIdx.x;
    int total = Bn * Nq * 256;
    if (tid >= total) return;
    int d = tid & 31;
    int h = (tid >> 5) & 7;
    int n = (tid >> 8) % Nq;
    int b = tid / (256 * Nq);

    const float* rw = refw + (size_t)b * refw_bstride + (size_t)n * 4;
    float cx = rw[0], cy = rw[1], bw = rw[2], bh = rw[3];
    const float* at = att + ((size_t)(b * Nq + n) * AA) + h * 16;

    float acc = 0.f;
    int start = 0;
    const int HS[4] = {64, 32, 16, 8};
#pragma unroll
    for (int l = 0; l < 4; ++l) {
        const int Hl = HS[l], Wl = HS[l];
#pragma unroll
        for (int p = 0; p < 4; ++p) {
            float gx = (p & 1)  ? 0.25f : -0.25f;
            float gy = (p >> 1) ? 0.25f : -0.25f;
            float px = cx + gx * bw, py = cy + gy * bh;
            float x = px * Wl - 0.5f, y = py * Hl - 0.5f;
            float x0f = floorf(x), y0f = floorf(y);
            float wx = x - x0f, wy = y - y0f;
            int x0 = (int)x0f, y0 = (int)y0f;
            float s = 0.f;
#pragma unroll
            for (int cc = 0; cc < 4; ++cc) {
                int cxi = cc & 1, cyi = cc >> 1;
                int xi = x0 + cxi, yi = y0 + cyi;
                if (xi >= 0 && xi < Wl && yi >= 0 && yi < Hl) {
                    float cw = (cxi ? wx : 1.f - wx) * (cyi ? wy : 1.f - wy);
                    s += cw * V[(((size_t)b * S + start + yi * Wl + xi) * 256) + h * 32 + d];
                }
            }
            acc += at[l * 4 + p] * s;
        }
        start += Hl * Wl;
    }
    out[(size_t)(b * Nq + n) * 256 + h * 32 + d] = acc;
}

// ---------------- classification logits + validity mask ----------------
__global__ void k_logits(const float* __restrict__ X, const float* __restrict__ clsw,
                         const float* __restrict__ clsb, const float* __restrict__ refw,
                         float* __restrict__ logits)
{
    int t = blockIdx.x * blockDim.x + threadIdx.x;
    if (t >= Bn * S) return;
    int s = t % S;
    const float* x = X + (size_t)t * 256;
    float acc = clsb[0];
    for (int i = 0; i < 256; ++i) acc += x[i] * clsw[i];
    float cx = refw[s*4], cy = refw[s*4+1];
    bool valid = (cx > 0.01f) && (cx < 0.99f) && (cy > 0.01f) && (cy < 0.99f);
    logits[t] = valid ? acc : -65504.f;
}

// ---------------- iterative top-k (matches jax.lax.top_k ordering) ----------------
__global__ void k_topk(const float* __restrict__ logits, float* __restrict__ scratch,
                       int* __restrict__ idx)
{
    __shared__ float bv[256];
    __shared__ int   bi[256];
    const int b = blockIdx.x;
    const int t = threadIdx.x;
    for (int i = t; i < S; i += 256) scratch[b * S + i] = logits[b * S + i];
    __syncthreads();
    for (int q = 0; q < NQ; ++q) {
        float best = -3.4e38f; int besti = 0x7fffffff;
        for (int i = t; i < S; i += 256) {
            float v = scratch[b * S + i];
            if (v > best || (v == best && i < besti)) { best = v; besti = i; }
        }
        bv[t] = best; bi[t] = besti; __syncthreads();
        for (int s2 = 128; s2 > 0; s2 >>= 1) {
            if (t < s2) {
                if (bv[t+s2] > bv[t] || (bv[t+s2] == bv[t] && bi[t+s2] < bi[t])) {
                    bv[t] = bv[t+s2]; bi[t] = bi[t+s2];
                }
            }
            __syncthreads();
        }
        if (t == 0) { idx[b * NQ + q] = bi[0]; scratch[b * S + bi[0]] = -3.4e38f; }
        __syncthreads();
    }
}

// ---------------- gather selected queries + ref windows ----------------
__global__ void k_gather(const float* __restrict__ X, const float* __restrict__ refw,
                         const int* __restrict__ idx, float* __restrict__ qemb,
                         float* __restrict__ refsel)
{
    int t = blockIdx.x * blockDim.x + threadIdx.x;
    if (t >= Bn * NQ * 256) return;
    int d = t & 255;
    int q = (t >> 8) % NQ;
    int b = t / (256 * NQ);
    int id = idx[b * NQ + q];
    qemb[(size_t)(b * NQ + q) * 256 + d] = X[((size_t)b * S + id) * 256 + d];
    if (d < 4) refsel[(size_t)(b * NQ + q) * 4 + d] = refw[(size_t)id * 4 + d];
}

// ---------------- bbox head final: h2 @ bb_w3 + b3 + inv_sigmoid, then sigmoid ----
__global__ void k_bbfinal(const float* __restrict__ h2, const float* __restrict__ w3,
                          const float* __restrict__ b3, const float* __restrict__ refsel,
                          float* __restrict__ outref)
{
    int t = blockIdx.x * blockDim.x + threadIdx.x;
    if (t >= Bn * NQ * 4) return;
    int j = t & 3;
    int row = t >> 2;
    const float* h = h2 + (size_t)row * 256;
    float acc = b3[j];
    for (int i = 0; i < 256; ++i) acc += h[i] * w3[i * 4 + j];
    float r = refsel[t];
    r = fminf(fmaxf(r, 1e-5f), 1.f - 1e-5f);
    acc += logf(r / (1.f - r));
    outref[t] = 1.f / (1.f + expf(-acc));
}

// ---------------- sine positional embedding of out_ref ----------------
__global__ void k_qpos(const float* __restrict__ outref, float* __restrict__ qpos)
{
    int t = blockIdx.x * blockDim.x + threadIdx.x;
    if (t >= Bn * NQ * 256) return;
    int o = t & 255;
    int row = t >> 8;
    int c  = o >> 7;
    int m  = (o & 127) >> 1;
    int sn = o & 1;
    float dimt = powf(10000.f, (float)m / 64.f);
    const float TWO_PI = 6.283185307179586f;
    float t1 = outref[(size_t)row * 4 + c]     * TWO_PI / dimt;
    float t2 = outref[(size_t)row * 4 + 2 + c] * TWO_PI / dimt;
    qpos[t] = (sn ? cosf(t1) : sinf(t1)) + (sn ? cosf(t2) : sinf(t2));
}

// ---------------- decoder self-attention (300x300 per head) ----------------
// QK:[Bn,NQ,512] (q cols 0..255, k cols 256..511)  Vp:[Bn,NQ,256]  O:[Bn,NQ,256]
__global__ void k_selfattn(const float* __restrict__ QK, const float* __restrict__ Vp,
                           float* __restrict__ O)
{
    __shared__ float sc[NQ];
    __shared__ float sum_s;
    int id = blockIdx.x;                 // Bn*NH*NQ
    int qi = id % NQ;
    int h  = (id / NQ) % NH;
    int b  = id / (NQ * NH);
    const float* qv = QK + ((size_t)(b * NQ + qi) * 512) + h * 32;
    const float scale = rsqrtf(32.f);
    for (int kk = threadIdx.x; kk < NQ; kk += blockDim.x) {
        const float* kv = QK + ((size_t)(b * NQ + kk) * 512) + 256 + h * 32;
        float d = 0.f;
        for (int e = 0; e < 32; ++e) d += qv[e] * kv[e];
        sc[kk] = d * scale;
    }
    __syncthreads();
    if (threadIdx.x == 0) {
        float mx = sc[0];
        for (int i = 1; i < NQ; ++i) mx = fmaxf(mx, sc[i]);
        float s = 0.f;
        for (int i = 0; i < NQ; ++i) { sc[i] = expf(sc[i] - mx); s += sc[i]; }
        sum_s = s;
    }
    __syncthreads();
    if (threadIdx.x < 32) {
        int d = threadIdx.x;
        float acc = 0.f;
        for (int kk = 0; kk < NQ; ++kk)
            acc += sc[kk] * Vp[(size_t)(b * NQ + kk) * 256 + h * 32 + d];
        O[(size_t)(b * NQ + qi) * 256 + h * 32 + d] = acc / sum_s;
    }
}

// ======================================================================
extern "C" void kernel_launch(void* const* d_in, const int* in_sizes, int n_in,
                              void* d_out, int out_size, void* d_ws, size_t ws_size,
                              hipStream_t stream)
{
    // ---- inputs (setup_inputs order) ----
    const float* src   = (const float*)d_in[0];
    const float* pos   = (const float*)d_in[1];
    const float* eWv   = (const float*)d_in[2];
    const float* ebv   = (const float*)d_in[3];
    const float* eWa   = (const float*)d_in[4];
    const float* eba   = (const float*)d_in[5];
    const float* eWo   = (const float*)d_in[6];
    const float* ebo   = (const float*)d_in[7];
    const float* eW1   = (const float*)d_in[8];
    const float* eb1   = (const float*)d_in[9];
    const float* eW2   = (const float*)d_in[10];
    const float* eb2   = (const float*)d_in[11];
    const float* en1g  = (const float*)d_in[12];
    const float* en1b  = (const float*)d_in[13];
    const float* en2g  = (const float*)d_in[14];
    const float* en2b  = (const float*)d_in[15];
    const float* dWqkv = (const float*)d_in[16];
    const float* dbqkv = (const float*)d_in[17];
    const float* dWso  = (const float*)d_in[18];
    const float* dbso  = (const float*)d_in[19];
    const float* dWv   = (const float*)d_in[20];
    const float* dbv   = (const float*)d_in[21];
    const float* dWa   = (const float*)d_in[22];
    const float* dba   = (const float*)d_in[23];
    const float* dWo   = (const float*)d_in[24];
    const float* dbo   = (const float*)d_in[25];
    const float* dW1   = (const float*)d_in[26];
    const float* db1   = (const float*)d_in[27];
    const float* dW2   = (const float*)d_in[28];
    const float* db2   = (const float*)d_in[29];
    const float* dn1g  = (const float*)d_in[30];
    const float* dn1b  = (const float*)d_in[31];
    const float* dn2g  = (const float*)d_in[32];
    const float* dn2b  = (const float*)d_in[33];
    const float* dn3g  = (const float*)d_in[34];
    const float* dn3b  = (const float*)d_in[35];
    const float* cls_w = (const float*)d_in[36];
    const float* cls_b = (const float*)d_in[37];
    const float* bb_w1 = (const float*)d_in[38];
    const float* bb_b1 = (const float*)d_in[39];
    const float* bb_w2 = (const float*)d_in[40];
    const float* bb_b2 = (const float*)d_in[41];
    const float* bb_w3 = (const float*)d_in[42];
    const float* bb_b3 = (const float*)d_in[43];
    const float* el_w  = (const float*)d_in[44];
    const float* el_b  = (const float*)d_in[45];
    const float* el_g  = (const float*)d_in[46];
    const float* el_bt = (const float*)d_in[47];

    // ---- workspace carve ----
    char* base = (char*)d_ws;
    size_t off = 0;
    auto alloc = [&](size_t nfloats) -> float* {
        float* p = (float*)(base + off);
        off += ((nfloats * sizeof(float) + 255) / 256) * 256;
        return p;
    };
    const size_t BSD = (size_t)Bn * S * D;      // 2,785,280
    const size_t BQD = (size_t)Bn * NQ * D;     // 153,600

    float* refw   = alloc((size_t)S * 4);
    float* X      = alloc(BSD);                 // encoder state
    float* Q      = alloc(BSD);                 // out + pos
    float* Vb     = alloc(BSD);                 // value projection (enc + dec cross)
    float* ATT    = alloc((size_t)Bn * S * AA);
    float* SAMP   = alloc(BSD);
    float* TMP    = alloc(BSD);
    float* F1     = alloc((size_t)Bn * S * FF); // also decoder FFN hidden
    float* logits = alloc((size_t)Bn * S);
    float* lscr   = alloc((size_t)Bn * S);
    int*   idx    = (int*)alloc((size_t)Bn * NQ);
    float* qemb   = alloc(BQD);
    float* refsel = alloc((size_t)Bn * NQ * 4);
    float* xq     = alloc(BQD);                 // decoder state
    float* qpos   = alloc(BQD);
    float* outref = alloc((size_t)Bn * NQ * 4);
    float* h1     = alloc(BQD);
    float* h2     = alloc(BQD);
    float* qin    = alloc(BQD);                 // x + qpos
    float* qkbuf  = alloc((size_t)Bn * NQ * 2 * D);
    float* vq     = alloc(BQD);
    float* sa     = alloc(BQD);
    float* attq   = alloc((size_t)Bn * NQ * AA);
    float* sampq  = alloc(BQD);
    float* tmpq   = alloc(BQD);

    auto gemm = [&](const float* Xp, const float* Wp, const float* bp, float* Yp,
                    int M, int N, int K, int wld, int relu) {
        dim3 g(N / 64, (M + 127) / 128), bl(256);
        hipLaunchKernelGGL(k_gemm, g, bl, 0, stream, Xp, Wp, bp, Yp, M, N, K, wld, relu);
    };

    const int MS = Bn * S;    // 10880 = 85 * 128
    const int MQ = Bn * NQ;   // 600

    // ---- static precompute ----
    hipLaunchKernelGGL(k_refwin, dim3((S + 255) / 256), dim3(256), 0, stream, refw);
    hipMemcpyAsync(X, src, BSD * sizeof(float), hipMemcpyDeviceToDevice, stream);

    // ================= encoder =================
    for (int i = 0; i < NE; ++i) {
        const float* Wv_i = eWv + (size_t)i * D * D;
        const float* Wa_i = eWa + (size_t)i * D * AA;
        const float* Wo_i = eWo + (size_t)i * D * D;
        const float* W1_i = eW1 + (size_t)i * D * FF;
        const float* W2_i = eW2 + (size_t)i * FF * D;

        hipLaunchKernelGGL(k_add, dim3((BSD + 255) / 256), dim3(256), 0, stream,
                           X, pos, Q, (int)BSD);
        gemm(X, Wv_i, ebv + (size_t)i * D, Vb, MS, D, D, D, 0);
        gemm(Q, Wa_i, eba + (size_t)i * AA, ATT, MS, AA, D, AA, 0);
        hipLaunchKernelGGL(k_softmax16, dim3((MS * NH + 255) / 256), dim3(256), 0, stream,
                           ATT, MS * NH);
        hipLaunchKernelGGL(k_boxsample, dim3((Bn * S * 256 + 255) / 256), dim3(256), 0, stream,
                           Vb, ATT, refw, 0, SAMP, S);
        gemm(SAMP, Wo_i, ebo + (size_t)i * D, TMP, MS, D, D, D, 0);
        hipLaunchKernelGGL(k_add_ln, dim3(MS), dim3(256), 0, stream,
                           X, TMP, en1g + (size_t)i * D, en1b + (size_t)i * D, X);
        gemm(X, W1_i, eb1 + (size_t)i * FF, F1, MS, FF, D, FF, 1);
        gemm(F1, W2_i, eb2 + (size_t)i * D, TMP, MS, D, FF, D, 0);
        hipLaunchKernelGGL(k_add_ln, dim3(MS), dim3(256), 0, stream,
                           X, TMP, en2g + (size_t)i * D, en2b + (size_t)i * D, X);
    }

    // ================= proposal selection =================
    hipLaunchKernelGGL(k_logits, dim3((MS + 255) / 256), dim3(256), 0, stream,
                       X, cls_w, cls_b, refw, logits);
    hipLaunchKernelGGL(k_topk, dim3(Bn), dim3(256), 0, stream, logits, lscr, idx);
    hipLaunchKernelGGL(k_gather, dim3((Bn * NQ * 256 + 255) / 256), dim3(256), 0, stream,
                       X, refw, idx, qemb, refsel);

    // tgt = LN(qemb @ el_w + el_b)
    gemm(qemb, el_w, el_b, tmpq, MQ, D, D, D, 0);
    hipLaunchKernelGGL(k_add_ln, dim3(MQ), dim3(256), 0, stream,
                       (const float*)nullptr, tmpq, el_g, el_bt, xq);

    // bbox head -> out_ref
    gemm(qemb, bb_w1, bb_b1, h1, MQ, D, D, D, 1);
    gemm(h1,   bb_w2, bb_b2, h2, MQ, D, D, D, 1);
    hipLaunchKernelGGL(k_bbfinal, dim3((Bn * NQ * 4 + 255) / 256), dim3(256), 0, stream,
                       h2, bb_w3, bb_b3, refsel, outref);
    hipLaunchKernelGGL(k_qpos, dim3((Bn * NQ * 256 + 255) / 256), dim3(256), 0, stream,
                       outref, qpos);

    // ================= decoder =================
    float* outp = (float*)d_out;
    for (int i = 0; i < ND; ++i) {
        const float* Wqkv_i = dWqkv + (size_t)i * D * 3 * D;
        const float* bqkv_i = dbqkv + (size_t)i * 3 * D;
        const float* Wso_i  = dWso + (size_t)i * D * D;
        const float* Wv_i   = dWv  + (size_t)i * D * D;
        const float* Wa_i   = dWa  + (size_t)i * D * AA;
        const float* Wo_i   = dWo  + (size_t)i * D * D;
        const float* W1_i   = dW1  + (size_t)i * D * FF;
        const float* W2_i   = dW2  + (size_t)i * FF * D;

        // --- self attention ---
        hipLaunchKernelGGL(k_add, dim3(((int)BQD + 255) / 256), dim3(256), 0, stream,
                           xq, qpos, qin, (int)BQD);
        gemm(qin, Wqkv_i,       bqkv_i,         qkbuf, MQ, 2 * D, D, 3 * D, 0); // Q|K
        gemm(xq,  Wqkv_i + 2*D, bqkv_i + 2 * D, vq,    MQ, D,     D, 3 * D, 0); // V
        hipLaunchKernelGGL(k_selfattn, dim3(Bn * NH * NQ), dim3(128), 0, stream,
                           qkbuf, vq, sa);
        gemm(sa, Wso_i, dbso + (size_t)i * D, tmpq, MQ, D, D, D, 0);
        hipLaunchKernelGGL(k_add_ln, dim3(MQ), dim3(256), 0, stream,
                           xq, tmpq, dn1g + (size_t)i * D, dn1b + (size_t)i * D, xq);

        // --- cross box attention (memory = encoder out X) ---
        hipLaunchKernelGGL(k_add, dim3(((int)BQD + 255) / 256), dim3(256), 0, stream,
                           xq, qpos, qin, (int)BQD);
        gemm(X,   Wv_i, dbv + (size_t)i * D,  Vb,   MS, D,  D, D,  0);
        gemm(qin, Wa_i, dba + (size_t)i * AA, attq, MQ, AA, D, AA, 0);
        hipLaunchKernelGGL(k_softmax16, dim3((MQ * NH + 255) / 256), dim3(256), 0, stream,
                           attq, MQ * NH);
        hipLaunchKernelGGL(k_boxsample, dim3((Bn * NQ * 256 + 255) / 256), dim3(256), 0, stream,
                           Vb, attq, outref, NQ * 4, sampq, NQ);
        gemm(sampq, Wo_i, dbo + (size_t)i * D, tmpq, MQ, D, D, D, 0);
        hipLaunchKernelGGL(k_add_ln, dim3(MQ), dim3(256), 0, stream,
                           xq, tmpq, dn2g + (size_t)i * D, dn2b + (size_t)i * D, xq);

        // --- FFN ---
        gemm(xq, W1_i, db1 + (size_t)i * FF, F1,   MQ, FF, D,  FF, 1);
        gemm(F1, W2_i, db2 + (size_t)i * D,  tmpq, MQ, D,  FF, D,  0);
        hipLaunchKernelGGL(k_add_ln, dim3(MQ), dim3(256), 0, stream,
                           xq, tmpq, dn3g + (size_t)i * D, dn3b + (size_t)i * D, xq);

        hipMemcpyAsync(outp + (size_t)i * BQD, xq, BQD * sizeof(float),
                       hipMemcpyDeviceToDevice, stream);
    }

    (void)in_sizes; (void)n_in; (void)out_size; (void)ws_size;
}
